// BasicBlock_72894184948228
// MI455X (gfx1250) — matrix operand order repaired and checked by reading, not compiled, not run on hardware
//
#include <hip/hip_runtime.h>
#include <hip/hip_bf16.h>

typedef __attribute__((ext_vector_type(16))) _Float16 v16h;
typedef __attribute__((ext_vector_type(8)))  float    v8f;

#define CCH 64
#define HDIM 56
#define WDIM 56
#define KTOT 576          // 64 channels * 9 taps, k = i*9 + tap
#define AS_STRIDE 584     // padded halves per im2col row (16B aligned, breaks bank stride)
#define OS_STRIDE 68      // padded floats per output-staging row

// ---------------- prep: binarize weights to f16, fold BN/shift constants ----------------
__global__ __launch_bounds__(256) void prep_kernel(
    const float* __restrict__ w, const float* __restrict__ shift1,
    const float* __restrict__ shift2, const float* __restrict__ w1,
    const float* __restrict__ gamma, const float* __restrict__ beta,
    const float* __restrict__ mean, const float* __restrict__ var,
    _Float16* __restrict__ Wh, float* __restrict__ alpha, float* __restrict__ beta_eff)
{
    __shared__ float s1[576], s2[576];
    int tid = threadIdx.x;
    // Bt[o][k] = sign(weight[o][i][kh][kw])  (flat OIHW is already o-major, k-contiguous)
    for (int idx = tid; idx < CCH * KTOT; idx += 256) {
        float v = w[idx];
        float s = (v > 0.f) ? 1.f : ((v < 0.f) ? -1.f : 0.f);
        Wh[idx] = (_Float16)s;
    }
    // per (o, tap) dot of sign(w) with each shift vector
    for (int idx = tid; idx < 576; idx += 256) {
        int o = idx / 9, tap = idx - 9 * (idx / 9);
        float a1 = 0.f, a2 = 0.f;
        for (int i = 0; i < CCH; ++i) {
            float v = w[o * KTOT + i * 9 + tap];
            float s = (v > 0.f) ? 1.f : ((v < 0.f) ? -1.f : 0.f);
            a1 += s * shift1[i];
            a2 += s * shift2[i];
        }
        s1[idx] = a1; s2[idx] = a2;
    }
    __syncthreads();
    if (tid < CCH) {
        int o = tid;
        float inv = rsqrtf(var[o] + 1e-5f);
        float g   = gamma[o] * inv;
        float w1o = w1[o];
        alpha[o]  = (1.f + w1o) * g;
        for (int hc = 0; hc < 3; ++hc)
            for (int wc = 0; wc < 3; ++wc) {
                int kh0 = (hc == 0) ? 1 : 0, kh1 = (hc == 2) ? 1 : 2;
                int kw0 = (wc == 0) ? 1 : 0, kw1 = (wc == 2) ? 1 : 2;
                float s = 0.f;
                for (int kh = kh0; kh <= kh1; ++kh)
                    for (int kw = kw0; kw <= kw1; ++kw) {
                        int tap = kh * 3 + kw;
                        s += s1[o * 9 + tap] + w1o * s2[o * 9 + tap];
                    }
                beta_eff[o * 9 + hc * 3 + wc] = (s - mean[o]) * g + beta[o];
            }
    }
}

// ---------------- main: implicit-GEMM binarized conv3x3 + fused epilogue ----------------
__global__ __launch_bounds__(256) void bconv_kernel(
    const float* __restrict__ x, const _Float16* __restrict__ Wh,
    const float* __restrict__ alpha, const float* __restrict__ beta_eff,
    float* __restrict__ out)
{
    __shared__ __align__(16) unsigned char smem[CCH * AS_STRIDE * 2];
    _Float16* As = (_Float16*)smem;          // im2col tile [m=64][k=576] (padded rows)
    float*    Os = (float*)smem;             // reused: accum staging [o=64][m] (padded)

    int blk = blockIdx.x;
    int n = blk / HDIM, h = blk - HDIM * (blk / HDIM);
    int tid = threadIdx.x;
    const float* xn = x + (size_t)n * (CCH * HDIM * WDIM);

    // ---- stage im2col A tile (f32 -> f16), coalesced over m ----
    for (int idx = tid; idx < CCH * KTOT; idx += 256) {
        int m = idx & 63;
        int k = idx >> 6;
        int i = k / 9;
        int tap = k - i * 9;
        int dh = tap / 3 - 1;
        int dw = tap - (tap / 3) * 3 - 1;
        int hh = h + dh, ww = m + dw;
        float v = 0.f;
        if (m < WDIM && (unsigned)hh < (unsigned)HDIM && (unsigned)ww < (unsigned)WDIM)
            v = xn[(i * HDIM + hh) * WDIM + ww];
        As[m * AS_STRIDE + k] = (_Float16)v;
    }
    __syncthreads();

    // ---- per-wave GEMM: 1 M-tile x 2 O-tiles, K-loop of 18 WMMAs each ----
    int lane = tid & 31, wid = tid >> 5;
    int m_tile = wid & 3;          // 4 M-tiles of 16
    int o_pair = wid >> 1 >> 1;    // 0..1 -> O-tiles {0,1} or {2,3}
    int half = lane >> 4;          // K-half select per ISA fragment layout
    int l15  = lane & 15;

    const _Float16* aP  = As + (m_tile * 16 + l15) * AS_STRIDE + half * 8;
    int o0 = o_pair * 32 + l15;
    const _Float16* b0P = Wh + o0 * KTOT + half * 8;
    const _Float16* b1P = b0P + 16 * KTOT;

    v8f c0 = {}; v8f c1 = {};
    for (int kb = 0; kb < KTOT; kb += 32) {
        union { v16h h; uint4 u[2]; } a, b0, b1;
        a.u[0]  = *(const uint4*)(aP + kb);        // k {0..7}/{8..15}
        a.u[1]  = *(const uint4*)(aP + kb + 16);   // k {16..23}/{24..31}
        b0.u[0] = *(const uint4*)(b0P + kb);       // k {0..7}/{8..15}
        b0.u[1] = *(const uint4*)(b0P + kb + 16);  // k {16..23}/{24..31}
        b1.u[0] = *(const uint4*)(b1P + kb);
        b1.u[1] = *(const uint4*)(b1P + kb + 16);
        c0 = __builtin_amdgcn_wmma_f32_16x16x32_f16(false, a.h, false, b0.h,
                                                    (short)0, c0, false, false);
        c1 = __builtin_amdgcn_wmma_f32_16x16x32_f16(false, a.h, false, b1.h,
                                                    (short)0, c1, false, false);
    }
    __syncthreads();   // everyone done reading As before aliasing it with Os

    // ---- stage accumulators to LDS as [o][m] (C/D layout: M = r + half*8, N = l15) ----
    int mbase = m_tile * 16 + half * 8;
#pragma unroll
    for (int r = 0; r < 8; ++r) {
        Os[o0 * OS_STRIDE + mbase + r]        = c0[r];
        Os[(o0 + 16) * OS_STRIDE + mbase + r] = c1[r];
    }
    __syncthreads();

    // ---- fused epilogue: alpha*acc + beta_edge + residual, clamp; coalesced over w ----
    int hc = (h == 0) ? 0 : ((h == HDIM - 1) ? 2 : 1);
    const float* xrn  = xn + h * WDIM;
    float*       outn = out + (size_t)n * (CCH * HDIM * WDIM) + h * WDIM;
    for (int idx = tid; idx < CCH * WDIM; idx += 256) {
        int o = idx / WDIM;
        int w = idx - o * WDIM;
        int wc = (w == 0) ? 0 : ((w == WDIM - 1) ? 2 : 1);
        float acc = Os[o * OS_STRIDE + w];
        float val = alpha[o] * acc + beta_eff[o * 9 + hc * 3 + wc] + xrn[o * (HDIM * WDIM) + w];
        val = fminf(1.f, fmaxf(-1.f, val));
        outn[o * (HDIM * WDIM) + w] = val;
    }
}

extern "C" void kernel_launch(void* const* d_in, const int* in_sizes, int n_in,
                              void* d_out, int out_size, void* d_ws, size_t ws_size,
                              hipStream_t stream) {
    const float* x      = (const float*)d_in[0];
    const float* shift1 = (const float*)d_in[1];
    const float* shift2 = (const float*)d_in[2];
    const float* weight = (const float*)d_in[3];
    const float* w1     = (const float*)d_in[4];
    const float* gamma  = (const float*)d_in[5];
    const float* beta   = (const float*)d_in[6];
    const float* rmean  = (const float*)d_in[7];
    const float* rvar   = (const float*)d_in[8];

    _Float16* Wh    = (_Float16*)d_ws;                    // 64*576*2 = 73728 B
    float* alpha    = (float*)((char*)d_ws + 73728);      // 256 B
    float* beta_eff = alpha + 64;                         // 64*9*4 = 2304 B

    prep_kernel<<<1, 256, 0, stream>>>(weight, shift1, shift2, w1, gamma, beta,
                                       rmean, rvar, Wh, alpha, beta_eff);

    int nblocks = 64 * HDIM;  // one (n, h) row per block
    bconv_kernel<<<nblocks, 256, 0, stream>>>(x, Wh, alpha, beta_eff, (float*)d_out);
}
